// GNN_GINSAGE_Model_67980742361722
// MI455X (gfx1250) — matrix operand order repaired and checked by reading, not compiled, not run on hardware
//
#include <hip/hip_runtime.h>
#include <hip/hip_bf16.h>
#include <math.h>

// ---------------------------------------------------------------------------
// GNN GIN+SAGE pipeline for MI455X (gfx1250), wave32 + WMMA bf16.
// Operands are pre-packed into WMMA fragment layout (bf16); the GEMM inner
// loop is software-pipelined b128 loads + v_wmma_f32_16x16x32_bf16.
// ---------------------------------------------------------------------------

typedef __attribute__((ext_vector_type(16))) __bf16 v16bf;
typedef __attribute__((ext_vector_type(8)))  float  v8f;

#define H_DIM   256
#define F_DIM   64
#define CAT_DIM 320

enum { MODE_SUM = 0, MODE_SAGE = 1 };

__device__ __forceinline__ __bf16 f32_to_bf16(float f) {
  unsigned int u = __builtin_bit_cast(unsigned int, f);
  u += 0x7fffu + ((u >> 16) & 1u);              // round-to-nearest-even
  unsigned short h = (unsigned short)(u >> 16);
  return __builtin_bit_cast(__bf16, h);
}

__device__ __forceinline__ v8f wmma_bf16(v16bf a, v16bf b, v8f c) {
  return __builtin_amdgcn_wmma_f32_16x16x32_bf16(false, a, false, b, (short)0, c,
                                                 false, false);
}

// ---------------------------- utility kernels ------------------------------

__global__ void k_zero(float* __restrict__ p, long long n) {
  long long i = (long long)blockIdx.x * blockDim.x + threadIdx.x;
  if (i < n) p[i] = 0.0f;
}

__global__ void k_degree(const int* __restrict__ dst, float* __restrict__ deg, int E) {
  int e = blockIdx.x * blockDim.x + threadIdx.x;
  if (e < E) atomicAdd(&deg[dst[e]], 1.0f);
}

__global__ void k_agg_scalar(const int* __restrict__ src, const int* __restrict__ dst,
                             const float* __restrict__ x1, float* __restrict__ agg, int E) {
  int e = blockIdx.x * blockDim.x + threadIdx.x;
  if (e < E) atomicAdd(&agg[dst[e]], x1[src[e]]);
}

// h1[n,j] = tanh((x1[n] + agg[n]) * w0[j] + b0[j])   (GIN layer 0: K==1 GEMM)
__global__ void k_gin0(const float* __restrict__ x1, const float* __restrict__ agg,
                       const float* __restrict__ w0, const float* __restrict__ b0,
                       float* __restrict__ h1, int n_) {
  int t = blockIdx.x * blockDim.x + threadIdx.x;
  if (t >= n_ * H_DIM) return;
  int n = t >> 8, j = t & 255;
  float s = x1[n] + agg[n];
  h1[t] = tanhf(fmaf(s, w0[j], b0[j]));
}

// copy x into the first F_DIM columns of the concat buffer (stride CAT_DIM)
__global__ void k_copy_x(const float* __restrict__ x, float* __restrict__ bufA, int n_) {
  int t = blockIdx.x * blockDim.x + threadIdx.x;
  if (t >= n_ * F_DIM) return;
  int n = t >> 6, f = t & 63;
  bufA[(size_t)n * CAT_DIM + f] = x[t];
}

// agg[dst] += feat[src], row width Kfeat (gridDim.y = Kfeat/4 chunks of 4 floats)
__global__ void k_scatter_rows(const int* __restrict__ src, const int* __restrict__ dst,
                               const float* __restrict__ feat, int ldf,
                               float* __restrict__ agg, int lda, int E) {
  int e = blockIdx.x * blockDim.x + threadIdx.x;
  if (e >= E) return;
  int c4 = blockIdx.y << 2;
  const float4 v = *(const float4*)(feat + (size_t)src[e] * ldf + c4);
  float* p = agg + (size_t)dst[e] * lda + c4;
  atomicAdd(p + 0, v.x);
  atomicAdd(p + 1, v.y);
  atomicAdd(p + 2, v.z);
  atomicAdd(p + 3, v.w);
}

// out[n,:] = h[n,:] @ W[256,2] + b   (tiny final linear, fp32 FMA)
__global__ void k_out_linear(const float* __restrict__ h, const float* __restrict__ w,
                             const float* __restrict__ b, float* __restrict__ out, int n_) {
  int n = blockIdx.x * blockDim.x + threadIdx.x;
  if (n >= n_) return;
  float a0 = b[0], a1 = b[1];
  const float* hr = h + (size_t)n * H_DIM;
#pragma unroll 8
  for (int k = 0; k < H_DIM; ++k) {
    float v = hr[k];
    a0 = fmaf(v, w[2 * k + 0], a0);
    a1 = fmaf(v, w[2 * k + 1], a1);
  }
  out[2 * n + 0] = a0;
  out[2 * n + 1] = a1;
}

// ------------------------- operand packing ---------------------------------
// Fragment record: one v16bf (32 B) per lane per 16x32 (A) / 32x16 (B) tile.
// A layout (ISA 7.12.2, 16-bit A 16x32): lane -> row (M = lane%16);
//   element i -> k = (i<8 ? i : 8+i) + 8*(lane/16).
// B layout: lane -> col (N = lane%16); element i -> k same as A.

// Pack A = fused activation matrix, bf16, index ((rt*KC + kc)*32 + lane).
template <int MODE>
__global__ void k_prepA(const float* __restrict__ s0, int ld0,
                        const float* __restrict__ s1, int ld1,
                        const float* __restrict__ deg,
                        int K1, int KC, v16bf* __restrict__ packA, int Mtiles) {
  int t = blockIdx.x * blockDim.x + threadIdx.x;    // (rt*KC + kc)*32 + lane
  if (t >= Mtiles * KC * 32) return;
  const int lane = t & 31;
  const int kc   = (t >> 5) % KC;
  const int rt   = (t >> 5) / KC;
  const int g = lane >> 4, lr = lane & 15;
  const int row = rt * 16 + lr;
  float invd = 1.0f;
  if (MODE == MODE_SAGE) invd = 1.0f / fmaxf(deg[row], 1.0f);
  v16bf a;
#pragma unroll
  for (int i = 0; i < 16; ++i) {
    const int kl = (i < 8) ? (i + 8 * g) : (8 + i + 8 * g);
    const int k  = kc * 32 + kl;
    float v;
    if (MODE == MODE_SUM) {
      v = s0[(size_t)row * ld0 + k] + s1[(size_t)row * ld1 + k];
    } else {
      v = (k < K1) ? s0[(size_t)row * ld0 + k] * invd
                   : s1[(size_t)row * ld1 + (k - K1)];
    }
    a[i] = f32_to_bf16(v);
  }
  packA[t] = a;
}

// Pack B = [B0;B1] (K-concat), bf16, index ((ct*KC + kc)*32 + lane).
__global__ void k_prepB(const float* __restrict__ B0, const float* __restrict__ B1,
                        int K1, int KC, v16bf* __restrict__ packB) {
  int t = blockIdx.x * blockDim.x + threadIdx.x;    // (ct*KC + kc)*32 + lane
  if (t >= (H_DIM / 16) * KC * 32) return;
  const int lane = t & 31;
  const int kc   = (t >> 5) % KC;
  const int ct   = (t >> 5) / KC;
  const int g = lane >> 4, lr = lane & 15;
  const int col = ct * 16 + lr;
  v16bf b;
#pragma unroll
  for (int i = 0; i < 16; ++i) {
    const int k = kc * 32 + ((i < 8) ? (i + 8 * g) : (8 + i + 8 * g));
    const float w = (k < K1) ? B0[(size_t)k * H_DIM + col]
                             : B1[(size_t)(k - K1) * H_DIM + col];
    b[i] = f32_to_bf16(w);
  }
  packB[t] = b;
}

// ---------------------------- WMMA GEMM ------------------------------------
// D = act(packA @ packB + bias). Wave -> 16x64 tile, 8 waves/block -> 32x256.
// Software-pipelined: chunk kc+1 fragment loads are issued before chunk kc's
// WMMAs so s_wait_loadcnt only covers loads one batch old.
__global__ __launch_bounds__(256) void k_gemm_packed(
    const v16bf* __restrict__ packA, const v16bf* __restrict__ packB,
    int KC, const float* __restrict__ bias,
    float* __restrict__ out, int ldo, int Mtiles, int act) {
  const int lane  = threadIdx.x & 31;
  const int wave  = threadIdx.x >> 5;
  const int waveM = wave >> 2;  // 0..1
  const int waveN = wave & 3;   // 0..3
  const int rt = blockIdx.x * 2 + waveM;
  if (rt >= Mtiles) return;                // wave-uniform: EXEC stays all-ones
  const int g = lane >> 4, lr = lane & 15;
  const int rowBase = rt * 16;
  const int colBase = waveN * 64;

  const v16bf* pa  = packA + (size_t)rt * KC * 32 + lane;
  const v16bf* pb0 = packB + (size_t)(waveN * 4 + 0) * KC * 32 + lane;
  const v16bf* pb1 = packB + (size_t)(waveN * 4 + 1) * KC * 32 + lane;
  const v16bf* pb2 = packB + (size_t)(waveN * 4 + 2) * KC * 32 + lane;
  const v16bf* pb3 = packB + (size_t)(waveN * 4 + 3) * KC * 32 + lane;

  v8f acc[4] = {};

  // Prologue: fragments for chunk 0.
  v16bf a  = pa[0];
  v16bf b0 = pb0[0], b1 = pb1[0], b2 = pb2[0], b3 = pb3[0];

#pragma unroll 2
  for (int kc = 0; kc + 1 < KC; ++kc) {
    // Issue next chunk's loads first (distinct registers -> partial waits).
    const size_t o = (size_t)(kc + 1) * 32;
    v16bf an  = pa[o];
    v16bf b0n = pb0[o], b1n = pb1[o], b2n = pb2[o], b3n = pb3[o];
    acc[0] = wmma_bf16(a, b0, acc[0]);
    acc[1] = wmma_bf16(a, b1, acc[1]);
    acc[2] = wmma_bf16(a, b2, acc[2]);
    acc[3] = wmma_bf16(a, b3, acc[3]);
    a = an; b0 = b0n; b1 = b1n; b2 = b2n; b3 = b3n;
  }
  // Epilogue chunk.
  acc[0] = wmma_bf16(a, b0, acc[0]);
  acc[1] = wmma_bf16(a, b1, acc[1]);
  acc[2] = wmma_bf16(a, b2, acc[2]);
  acc[3] = wmma_bf16(a, b3, acc[3]);

  // C/D layout: lanes 0-15 VGPR r -> M=r,N=lane; lanes 16-31 -> M=8+r,N=lane-16
#pragma unroll
  for (int n = 0; n < 4; ++n) {
    const int col = colBase + n * 16 + lr;
    const float bi = bias[col];
#pragma unroll
    for (int r = 0; r < 8; ++r) {
      const int row = rowBase + g * 8 + r;
      float v = acc[n][r] + bi;
      if (act == 1)      v = tanhf(v);
      else if (act == 2) v = fmaxf(v, 0.0f);
      out[(size_t)row * ldo + col] = v;
    }
  }
}

// ---------------------------- launch ---------------------------------------

extern "C" void kernel_launch(void* const* d_in, const int* in_sizes, int n_in,
                              void* d_out, int out_size, void* d_ws, size_t ws_size,
                              hipStream_t stream) {
  const float* x      = (const float*)d_in[0];
  const float* x1     = (const float*)d_in[1];
  const int*   ei     = (const int*)d_in[2];
  const float* gin_w0 = (const float*)d_in[3];
  const float* gin_b0 = (const float*)d_in[4];
  const float* gin_w1 = (const float*)d_in[5];
  const float* gin_b1 = (const float*)d_in[6];
  const float* wl0    = (const float*)d_in[7];
  const float* bl0    = (const float*)d_in[8];
  const float* wr0    = (const float*)d_in[9];
  const float* wl1    = (const float*)d_in[10];
  const float* bl1    = (const float*)d_in[11];
  const float* wr1    = (const float*)d_in[12];
  const float* lw     = (const float*)d_in[13];
  const float* lb     = (const float*)d_in[14];
  float* out = (float*)d_out;

  const int N = in_sizes[1];          // x1 is (N,1)
  const int E = in_sizes[2] / 2;      // edge_index is (2,E)
  const int* srcI = ei;
  const int* dstI = ei + E;
  const int Mtiles = (N + 15) / 16;   // N = 50000 -> 3125 exact

  // Workspace (floats): deg | bufA | bufB | bufC (N*320 each) | packA | packB
  float* W    = (float*)d_ws;
  size_t nAl  = ((size_t)N + 255) & ~(size_t)255;
  float* deg  = W;
  float* bufA = W + nAl;                       // concat [x | h1], later h3
  float* bufB = bufA + (size_t)N * CAT_DIM;    // aggregation scratch
  float* bufC = bufB + (size_t)N * CAT_DIM;    // h1, then h2
  v16bf* packA = (v16bf*)(bufC + (size_t)N * CAT_DIM);       // <= N*640 bf16
  v16bf* packB = packA + (size_t)Mtiles * (2 * CAT_DIM / 32) * 32;

  const int TB = 256;
  auto cdiv = [](long long a, long long b) { return (int)((a + b - 1) / b); };
  const int gemmGrid = cdiv(Mtiles, 2);

  // degree (in-degree at dst)
  k_zero<<<cdiv(N, TB), TB, 0, stream>>>(deg, N);
  k_degree<<<cdiv(E, TB), TB, 0, stream>>>(dstI, deg, E);

  // ---- GIN layer 0 (K==1, scalar agg + elementwise) ----
  k_zero<<<cdiv(N, TB), TB, 0, stream>>>(bufB, N);
  k_agg_scalar<<<cdiv(E, TB), TB, 0, stream>>>(srcI, dstI, x1, bufB, E);
  k_gin0<<<cdiv((long long)N * H_DIM, TB), TB, 0, stream>>>(x1, bufB, gin_w0, gin_b0, bufC, N);

  // ---- GIN layer 1: (h1 + agg(h1)) @ W1 + b1, tanh -> bufA[:,64:320] ----
  k_zero<<<cdiv((long long)N * H_DIM, TB), TB, 0, stream>>>(bufB, (long long)N * H_DIM);
  {
    dim3 grid(cdiv(E, TB), H_DIM / 4);
    k_scatter_rows<<<grid, TB, 0, stream>>>(srcI, dstI, bufC, H_DIM, bufB, H_DIM, E);
  }
  k_copy_x<<<cdiv((long long)N * F_DIM, TB), TB, 0, stream>>>(x, bufA, N);
  {
    const int KC = H_DIM / 32;  // 8
    k_prepB<<<cdiv((H_DIM / 16) * KC * 32, TB), TB, 0, stream>>>(gin_w1, gin_w1, H_DIM, KC, packB);
    k_prepA<MODE_SUM><<<cdiv((long long)Mtiles * KC * 32, TB), TB, 0, stream>>>(
        bufC, H_DIM, bufB, H_DIM, deg, H_DIM, KC, packA, Mtiles);
    k_gemm_packed<<<gemmGrid, 256, 0, stream>>>(packA, packB, KC, gin_b1,
                                                bufA + F_DIM, CAT_DIM, Mtiles, /*tanh*/ 1);
  }

  // ---- SAGE 0: [mean(concat)|concat] @ [Wl0;Wr0] + bl0, relu -> bufC ----
  k_zero<<<cdiv((long long)N * CAT_DIM, TB), TB, 0, stream>>>(bufB, (long long)N * CAT_DIM);
  {
    dim3 grid(cdiv(E, TB), CAT_DIM / 4);
    k_scatter_rows<<<grid, TB, 0, stream>>>(srcI, dstI, bufA, CAT_DIM, bufB, CAT_DIM, E);
  }
  {
    const int KC = 2 * CAT_DIM / 32;  // 20
    k_prepB<<<cdiv((H_DIM / 16) * KC * 32, TB), TB, 0, stream>>>(wl0, wr0, CAT_DIM, KC, packB);
    k_prepA<MODE_SAGE><<<cdiv((long long)Mtiles * KC * 32, TB), TB, 0, stream>>>(
        bufB, CAT_DIM, bufA, CAT_DIM, deg, CAT_DIM, KC, packA, Mtiles);
    k_gemm_packed<<<gemmGrid, 256, 0, stream>>>(packA, packB, KC, bl0,
                                                bufC, H_DIM, Mtiles, /*relu*/ 2);
  }

  // ---- SAGE 1: [mean(h2)|h2] @ [Wl1;Wr1] + bl1, relu -> bufA ----
  k_zero<<<cdiv((long long)N * H_DIM, TB), TB, 0, stream>>>(bufB, (long long)N * H_DIM);
  {
    dim3 grid(cdiv(E, TB), H_DIM / 4);
    k_scatter_rows<<<grid, TB, 0, stream>>>(srcI, dstI, bufC, H_DIM, bufB, H_DIM, E);
  }
  {
    const int KC = 2 * H_DIM / 32;  // 16
    k_prepB<<<cdiv((H_DIM / 16) * KC * 32, TB), TB, 0, stream>>>(wl1, wr1, H_DIM, KC, packB);
    k_prepA<MODE_SAGE><<<cdiv((long long)Mtiles * KC * 32, TB), TB, 0, stream>>>(
        bufB, H_DIM, bufC, H_DIM, deg, H_DIM, KC, packA, Mtiles);
    k_gemm_packed<<<gemmGrid, 256, 0, stream>>>(packA, packB, KC, bl1,
                                                bufA, H_DIM, Mtiles, /*relu*/ 2);
  }

  // ---- output linear [256 -> 2] ----
  k_out_linear<<<cdiv(N, TB), TB, 0, stream>>>(bufA, lw, lb, out, N);
}
